// BboxGrid_17446157156803
// MI455X (gfx1250) — matrix operand order repaired and checked by reading, not compiled, not run on hardware
//
#include <hip/hip_runtime.h>

// ---------------------------------------------------------------------------
// CDNA5 (gfx1250) types
// ---------------------------------------------------------------------------
typedef __attribute__((ext_vector_type(16))) __bf16 v16bf;
typedef __attribute__((ext_vector_type(8)))  float  v8f;
typedef unsigned int u32x4 __attribute__((ext_vector_type(4)));
typedef int          i32x8 __attribute__((ext_vector_type(8)));
typedef int          i32x4 __attribute__((ext_vector_type(4)));

union ABu { v16bf v; unsigned short u[16]; };

__device__ __forceinline__ unsigned short f32_to_bf16(float f) {
    unsigned int x = __float_as_uint(f);
    unsigned int r = (x + 0x7FFFu + ((x >> 16) & 1u)) >> 16;   // RNE
    return (unsigned short)r;
}
__device__ __forceinline__ float bf16_to_f32(unsigned short h) {
    return __uint_as_float(((unsigned int)h) << 16);
}
__device__ __forceinline__ float sigmoidf_(float x) {
    return 1.0f / (1.0f + __expf(-x));
}

// ---------------------------------------------------------------------------
// Weight packing: OIHW f32 [Cout][Cin][3][3] -> bf16 [Kpad][CP]
// K index = ci*9 + ky*3 + kx ; Kpad mult of 32, CP mult of 64, zero padded
// ---------------------------------------------------------------------------
__global__ void pack_w_kernel(const float* __restrict__ w,
                              unsigned short* __restrict__ wp,
                              int Cout, int Cin, int K, int Kpad, int CP)
{
    int i = blockIdx.x * blockDim.x + threadIdx.x;
    if (i >= Kpad * CP) return;
    int n  = i % CP;
    int kp = i / CP;
    unsigned short v = 0;
    if (kp < K && n < Cout) {
        int ci = kp / 9;
        int r  = kp - ci * 9;
        v = f32_to_bf16(w[(((long)n * Cin + ci) * 3 + (r / 3)) * 3 + (r % 3)]);
    }
    wp[i] = v;
}

// ---------------------------------------------------------------------------
// Implicit-GEMM 3x3 stride-2 pad-1 conv via V_WMMA_F32_16X16X32_BF16.
// Block = 256 threads (8 waves). Block tile 128(M) x 64(N).
// Wave (msub 0..3, nsub 0..1) owns a 32x32 output: 2 A-frags x 2 B-frags
// -> 4 WMMA per K-step. Weight tile fetched by the Tensor Data Mover
// (tensor_load_to_lds) with LDS row padding; A tile is manual im2col with
// the row decode hoisted out of the K loop and division-free K decode.
// ---------------------------------------------------------------------------
__global__ __launch_bounds__(256)
void conv_wmma_kernel(const void* __restrict__ xin, int x_is_f32,
                      const unsigned short* __restrict__ wp,
                      const float* __restrict__ bias,
                      void* __restrict__ yout, int y_is_f32,
                      int Nimg, int Cin, int Hin, int Win,
                      int Cout, int Hout, int Wout,
                      int K, int Kpad, int CP)
{
    __shared__ unsigned short As[128][33];   // 128 x 32 bf16 (+1 pad)
    __shared__ unsigned short Bs[32][66];    // TDM tile: 32 rows x (64 bf16 + 1dw pad)

    const int  tid   = threadIdx.x;
    const int  lane  = tid & 31;
    const int  wv    = tid >> 5;      // 0..7
    const int  msub  = wv >> 1;       // 0..3  -> M offset msub*32
    const int  nsub  = wv & 1;        // 0..1  -> N offset nsub*32
    const long M     = (long)Nimg * Hout * Wout;
    const long mBase = (long)blockIdx.x * 128;
    const int  nBase = blockIdx.y * 64;

    const float*          xf = (const float*)xin;
    const unsigned short* xb = (const unsigned short*)xin;

    // ---- per-thread A-tile position: one row, 16 columns (hoisted decode) ----
    const int  arow  = tid >> 1;            // 0..127
    const int  acol0 = (tid & 1) << 4;      // 0 or 16
    const long mRow  = mBase + arow;
    const bool mOK   = (mRow < M);
    int oy = 0, nn = 0, iyB = 0, ixB = 0;
    long ciBase = 0;
    if (mOK) {
        int  ox = (int)(mRow % Wout);
        long t  = mRow / Wout;
        oy      = (int)(t % Hout);
        nn      = (int)(t / Hout);
        iyB     = oy * 2 - 1;
        ixB     = ox * 2 - 1;
        ciBase  = (long)nn * Cin;
    }
    // division-free K decode state for the 16 columns this thread loads
    int ciA[16], r9A[16];
    #pragma unroll
    for (int i = 0; i < 16; ++i) {
        int kc  = acol0 + i;
        ciA[i]  = kc / 9;
        r9A[i]  = kc - ciA[i] * 9;
    }

    // ---- TDM descriptor pieces that do not change across K iterations ----
    const unsigned ldsB = (unsigned)(unsigned long long)(&Bs[0][0]);
    // group1: wg_mask=0, data_size=1(2B), pad_enable, pad_interval=4 (32 dwords),
    //         pad_amount=0 (1 dword); tensor CPxKpad; tile 64x32; stride CP
    const unsigned g1d0 = 0x00010000u | (1u << 20) | (4u << 22);
    const unsigned g1d1 = ((unsigned)CP & 0xFFFFu) << 16;
    const unsigned g1d2 = ((unsigned)CP >> 16) | (((unsigned)Kpad & 0xFFFFu) << 16);
    const unsigned g1d3 = ((unsigned)Kpad >> 16) | (64u << 16);
    const unsigned g1d4 = 32u;
    const unsigned g1d5 = (unsigned)CP;

    v8f acc00 = {}, acc01 = {}, acc10 = {}, acc11 = {};
    const int kIters = Kpad >> 5;

    for (int kt = 0; kt < kIters; ++kt) {
        // ---- weight tile 32(K) x 64(N) via Tensor Data Mover ----
        if (wv == 0) {
            unsigned long long gaddr =
                (unsigned long long)(const void*)&wp[(long)kt * 32 * CP + nBase];
            u32x4 g0;
            g0.x = 1u;                                  // count=1, user mode
            g0.y = ldsB;                                // lds_addr
            g0.z = (unsigned)gaddr;                     // global_addr[31:0]
            g0.w = ((unsigned)(gaddr >> 32) & 0x01FFFFFFu) | 0x80000000u; // [56:32]|type=2
            i32x8 g1;
            g1[0] = (int)g1d0; g1[1] = (int)g1d1; g1[2] = (int)g1d2; g1[3] = (int)g1d3;
            g1[4] = (int)g1d4; g1[5] = (int)g1d5; g1[6] = 0; g1[7] = 0;
            i32x4 gz4 = {0, 0, 0, 0};
            i32x8 gz8 = {0, 0, 0, 0, 0, 0, 0, 0};
            __builtin_amdgcn_tensor_load_to_lds(g0, g1, gz4, gz4, gz8, 0);
            __builtin_amdgcn_s_wait_tensorcnt(0);
        }

        // ---- A tile (128 rows x 32 K of im2col, f32->bf16) ----
        #pragma unroll
        for (int i = 0; i < 16; ++i) {
            int kg = kt * 32 + acol0 + i;
            unsigned short hv = 0;
            if (mOK && kg < K) {
                int r9 = r9A[i];
                int ky = r9 / 3;
                int kx = r9 - ky * 3;
                int iy = iyB + ky;
                int ix = ixB + kx;
                if ((unsigned)iy < (unsigned)Hin && (unsigned)ix < (unsigned)Win) {
                    long idx = ((ciBase + ciA[i]) * Hin + iy) * Win + ix;
                    hv = x_is_f32 ? f32_to_bf16(xf[idx]) : xb[idx];
                }
            }
            As[arow][acol0 + i] = hv;
        }
        // advance K decode by 32 (32 = 3*9 + 5)
        #pragma unroll
        for (int i = 0; i < 16; ++i) {
            int r9 = r9A[i] + 5;
            int ci = ciA[i] + 3;
            if (r9 >= 9) { r9 -= 9; ci += 1; }
            r9A[i] = r9; ciA[i] = ci;
        }
        __syncthreads();

        // ---- assemble fragments (same contiguous K order for A and B) ----
        // A elem e -> K = (lane>>4)*16 + e,               M = row
        // B elem e -> K = (lane>>4)*16 + e,               N = col
        ABu a0, a1, b0, b1;
        const int l15  = lane & 15;
        const int kHiA = (lane >> 4) << 3;
        const int kHiB = (lane >> 4) << 4;
        const int mr0  = (msub << 5) + l15;        // rows msub*32 + 0..15
        const int mr1  = mr0 + 16;                 // rows msub*32 + 16..31
        const int nc0  = (nsub << 5) + l15;        // cols nsub*32 + 0..15
        const int nc1  = nc0 + 16;
        #pragma unroll
        for (int e = 0; e < 16; ++e) {
            int kA = kHiB + e;
            a0.u[e] = As[mr0][kA];
            a1.u[e] = As[mr1][kA];
            int kB = kHiB + e;
            b0.u[e] = Bs[kB][nc0];
            b1.u[e] = Bs[kB][nc1];
        }
        acc00 = __builtin_amdgcn_wmma_f32_16x16x32_bf16(false, a0.v, false, b0.v, (short)0, acc00, false, false);
        acc01 = __builtin_amdgcn_wmma_f32_16x16x32_bf16(false, a0.v, false, b1.v, (short)0, acc01, false, false);
        acc10 = __builtin_amdgcn_wmma_f32_16x16x32_bf16(false, a1.v, false, b0.v, (short)0, acc10, false, false);
        acc11 = __builtin_amdgcn_wmma_f32_16x16x32_bf16(false, a1.v, false, b1.v, (short)0, acc11, false, false);
        __syncthreads();
    }

    // ---- store: C/D layout elem r -> M = r + (lane>>4)*8, N = lane&15 ----
    #pragma unroll
    for (int ai = 0; ai < 2; ++ai) {
        #pragma unroll
        for (int bi = 0; bi < 2; ++bi) {
            const v8f acc = (ai == 0) ? (bi == 0 ? acc00 : acc01)
                                      : (bi == 0 ? acc10 : acc11);
            #pragma unroll
            for (int r = 0; r < 8; ++r) {
                int  ml = r + ((lane >> 4) << 3);
                long m  = mBase + (msub << 5) + ai * 16 + ml;
                int  n  = nBase + (nsub << 5) + bi * 16 + (lane & 15);
                if (m < M && n < Cout) {
                    float v  = acc[r] + bias[n];
                    int  ox  = (int)(m % Wout);
                    long t   = m / Wout;
                    int  oyc = (int)(t % Hout);
                    int  nnc = (int)(t / Hout);
                    long oid = (((long)nnc * Cout + n) * Hout + oyc) * Wout + ox;
                    if (y_is_f32) ((float*)yout)[oid] = v;
                    else          ((unsigned short*)yout)[oid] = f32_to_bf16(v);
                }
            }
        }
    }
}

// ---------------------------------------------------------------------------
// Boxes from grid head g [16,5,32,32]
// ---------------------------------------------------------------------------
__global__ void bbox_kernel(const float* __restrict__ g, float* __restrict__ boxes)
{
    int i = blockIdx.x * blockDim.x + threadIdx.x;   // b*1024 + cell
    if (i >= 16 * 1024) return;
    int b    = i >> 10;
    int cell = i & 1023;
    int r    = cell >> 5;
    int c    = cell & 31;
    const float* gb = g + (long)b * 5 * 1024;
    float cy = 16.0f * (sigmoidf_(gb[0 * 1024 + cell]) + (float)r);
    float cx = 16.0f * (sigmoidf_(gb[1 * 1024 + cell]) + (float)c);
    float sh = 512.0f * sigmoidf_(gb[2 * 1024 + cell]);
    float sw = 512.0f * sigmoidf_(gb[3 * 1024 + cell]);
    float* o = boxes + (long)i * 4;
    o[0] = floorf(fmaxf(cy - sh * 0.5f, 0.0f));
    o[1] = floorf(fmaxf(cx - sw * 0.5f, 0.0f));
    o[2] = floorf(fminf(cy + sh * 0.5f, 511.0f));
    o[3] = floorf(fminf(cx + sw * 0.5f, 511.0f));
}

// ---------------------------------------------------------------------------
// Per-batch softmax over 1024 cells of channel 4
// ---------------------------------------------------------------------------
__global__ void softmax_kernel(const float* __restrict__ g,
                               float* __restrict__ scF,
                               float* __restrict__ outScores)
{
    __shared__ float red[256];
    int b = blockIdx.x, t = threadIdx.x;
    const float* gs = g + ((long)b * 5 + 4) * 1024;
    float mx = -1e30f;
    for (int i = t; i < 1024; i += 256) mx = fmaxf(mx, gs[i]);
    red[t] = mx; __syncthreads();
    for (int s = 128; s > 0; s >>= 1) { if (t < s) red[t] = fmaxf(red[t], red[t + s]); __syncthreads(); }
    mx = red[0]; __syncthreads();
    float sum = 0.0f;
    for (int i = t; i < 1024; i += 256) sum += __expf(gs[i] - mx);
    red[t] = sum; __syncthreads();
    for (int s = 128; s > 0; s >>= 1) { if (t < s) red[t] += red[t + s]; __syncthreads(); }
    float inv = 1.0f / red[0];
    for (int i = t; i < 1024; i += 256) {
        float v = __expf(gs[i] - mx) * inv;
        scF[b * 1024 + i]       = v;
        outScores[b * 1024 + i] = v;
    }
}

// ---------------------------------------------------------------------------
// RoIAlign-style bilinear crop+resize -> regions bf16 [16384,3,12,12]
// ---------------------------------------------------------------------------
__global__ void roi_kernel(const float* __restrict__ images,
                           const float* __restrict__ boxes,
                           unsigned short* __restrict__ regions)
{
    long i = (long)blockIdx.x * blockDim.x + threadIdx.x;
    if (i >= (long)16384 * 432) return;                 // 432 = 3*12*12
    int  rx  = (int)(i % 12);
    long t   = i / 12;
    int  ry  = (int)(t % 12);  t /= 12;
    int  ci  = (int)(t % 3);
    long roi = t / 3;
    int  b   = (int)(roi >> 10);
    const float* bx = boxes + roi * 4;
    float y0 = bx[0], x0 = bx[1], y1 = bx[2], x1 = bx[3];
    float bh = fmaxf(y1 - y0, 1.0f), bw = fmaxf(x1 - x0, 1.0f);
    float sy = y0 + ((ry + 0.5f) / 12.0f) * bh - 0.5f;
    float sx = x0 + ((rx + 0.5f) / 12.0f) * bw - 0.5f;
    float yf = floorf(sy), xf = floorf(sx);
    float wy = sy - yf,    wx = sx - xf;
    int y0i = min(max((int)yf, 0), 511);
    int y1i = min(max((int)yf + 1, 0), 511);
    int x0i = min(max((int)xf, 0), 511);
    int x1i = min(max((int)xf + 1, 0), 511);
    const float* img = images + (((long)b * 3 + ci) * 512) * 512;
    float v = img[(long)y0i * 512 + x0i] * (1.0f - wy) * (1.0f - wx)
            + img[(long)y0i * 512 + x1i] * (1.0f - wy) * wx
            + img[(long)y1i * 512 + x0i] * wy * (1.0f - wx)
            + img[(long)y1i * 512 + x1i] * wy * wx;
    regions[i] = f32_to_bf16(v);
}

// ---------------------------------------------------------------------------
// hidden[b,d] = sum_n scores[b,n] * lat[b,n,d]   (lat bf16 [16384,2304])
// ---------------------------------------------------------------------------
__global__ void hidden_kernel(const float* __restrict__ sc,
                              const unsigned short* __restrict__ lat,
                              float* __restrict__ hidden)
{
    int d = blockIdx.x * blockDim.x + threadIdx.x;
    int b = blockIdx.y;
    if (d >= 2304) return;
    const unsigned short* lb = lat + (long)b * 1024 * 2304;
    const float*          sb = sc  + (long)b * 1024;
    float acc = 0.0f;
    for (int n = 0; n < 1024; ++n)
        acc += sb[n] * bf16_to_f32(lb[(long)n * 2304 + d]);
    hidden[b * 2304 + d] = acc;
}

// ---------------------------------------------------------------------------
// logits = hidden @ ow + ob   ([16,2304] x [2304,80])
// ---------------------------------------------------------------------------
__global__ void logits_kernel(const float* __restrict__ hidden,
                              const float* __restrict__ ow,
                              const float* __restrict__ ob,
                              float* __restrict__ out)
{
    int n = threadIdx.x;
    int b = blockIdx.x;
    if (n >= 80) return;
    const float* h = hidden + b * 2304;
    float acc = ob[n];
    for (int k = 0; k < 2304; ++k) acc += h[k] * ow[k * 80 + n];
    out[b * 80 + n] = acc;
}

// ---------------------------------------------------------------------------
// Host-side orchestration
// ---------------------------------------------------------------------------
extern "C" void kernel_launch(void* const* d_in, const int* in_sizes, int n_in,
                              void* d_out, int out_size, void* d_ws, size_t ws_size,
                              hipStream_t stream)
{
    (void)in_sizes; (void)n_in; (void)out_size; (void)ws_size;
    const float* images = (const float*)d_in[0];
    const float* gw1 = (const float*)d_in[1];  const float* gb1 = (const float*)d_in[2];
    const float* gw2 = (const float*)d_in[3];  const float* gb2 = (const float*)d_in[4];
    const float* gw3 = (const float*)d_in[5];  const float* gb3 = (const float*)d_in[6];
    const float* gw4 = (const float*)d_in[7];  const float* gb4 = (const float*)d_in[8];
    const float* rw1 = (const float*)d_in[9];  const float* rb1 = (const float*)d_in[10];
    const float* rw2 = (const float*)d_in[11]; const float* rb2 = (const float*)d_in[12];
    const float* ow  = (const float*)d_in[13]; const float* ob  = (const float*)d_in[14];
    float* out = (float*)d_out;

    // ---- workspace layout (bytes) ----
    const size_t SZ_ACT1 = (size_t)16 * 128 * 256 * 256 * 2;  // 268.4 MB (reused for r1 out)
    const size_t SZ_ACT2 = (size_t)16 * 256 * 128 * 128 * 2;  // 134.2 MB
    const size_t SZ_ACT3 = (size_t)16 * 512 * 64 * 64 * 2;    // 67.1 MB (reused for regions)
    const size_t SZ_G    = (size_t)16 * 5 * 32 * 32 * 4;
    const size_t SZ_BOX  = (size_t)16 * 1024 * 4 * 4;
    const size_t SZ_SC   = (size_t)16 * 1024 * 4;
    const size_t SZ_HID  = (size_t)16 * 2304 * 4;
    const size_t SZ_WP   = (size_t)8 << 20;

    char* ws = (char*)d_ws;
    unsigned short* act1   = (unsigned short*)(ws);
    unsigned short* act2   = (unsigned short*)(ws + SZ_ACT1);
    unsigned short* act3   = (unsigned short*)(ws + SZ_ACT1 + SZ_ACT2);
    float*          g      = (float*)(ws + SZ_ACT1 + SZ_ACT2 + SZ_ACT3);
    float*          boxes  = (float*)((char*)g + SZ_G);
    float*          scF    = (float*)((char*)boxes + SZ_BOX);
    float*          hidden = (float*)((char*)scF + SZ_SC);
    char*           wpB    = (char*)hidden + SZ_HID;
    unsigned short* lat    = (unsigned short*)(wpB + SZ_WP);   // 16384*2304 bf16
    unsigned short* regions = act3;   // act3 free after conv4
    unsigned short* r1out   = act1;   // act1 free after conv2

    // packed weight sub-offsets (CP rounded to 64)
    unsigned short* w1p  = (unsigned short*)(wpB);                              //   32*128
    unsigned short* w2p  = (unsigned short*)(wpB + 8192);                       // 1152*256
    unsigned short* w3p  = (unsigned short*)(wpB + 8192 + 589824);              // 2304*512
    unsigned short* w4p  = (unsigned short*)(wpB + 8192 + 589824 + 2359296);    // 4608*64
    unsigned short* rw1p = (unsigned short*)(wpB + 8192 + 589824 + 2359296 + 589824);
    unsigned short* rw2p = (unsigned short*)(wpB + 8192 + 589824 + 2359296 + 589824 + 8192);

    auto pack = [&](const float* w, unsigned short* wp, int Cout, int Cin) {
        int K = Cin * 9, Kpad = (K + 31) & ~31, CP = (Cout + 63) & ~63;
        int tot = Kpad * CP;
        pack_w_kernel<<<(tot + 255) / 256, 256, 0, stream>>>(w, wp, Cout, Cin, K, Kpad, CP);
    };
    auto conv = [&](const void* x, int xf32, const unsigned short* wp, const float* bias,
                    void* y, int yf32, int Nimg, int Cin, int Hin, int Win,
                    int Cout, int Hout, int Wout) {
        int  K = Cin * 9, Kpad = (K + 31) & ~31, CP = (Cout + 63) & ~63;
        long M = (long)Nimg * Hout * Wout;
        dim3 grid((unsigned)((M + 127) / 128), (unsigned)(CP / 64));
        conv_wmma_kernel<<<grid, 256, 0, stream>>>(x, xf32, wp, bias, y, yf32,
                                                   Nimg, Cin, Hin, Win,
                                                   Cout, Hout, Wout, K, Kpad, CP);
    };

    // weight packing
    pack(gw1, w1p, 128, 3);
    pack(gw2, w2p, 256, 128);
    pack(gw3, w3p, 512, 256);
    pack(gw4, w4p, 5, 512);
    pack(rw1, rw1p, 128, 3);
    pack(rw2, rw2p, 256, 128);

    // backbone
    conv(images, 1, w1p, gb1, act1, 0, 16,   3, 512, 512, 128, 256, 256);
    conv(act1,   0, w2p, gb2, act2, 0, 16, 128, 256, 256, 256, 128, 128);
    conv(act2,   0, w3p, gb3, act3, 0, 16, 256, 128, 128, 512,  64,  64);
    conv(act3,   0, w4p, gb4, g,    1, 16, 512,  64,  64,   5,  32,  32);

    // boxes + scores
    bbox_kernel<<<(16 * 1024 + 255) / 256, 256, 0, stream>>>(g, boxes);
    softmax_kernel<<<16, 256, 0, stream>>>(g, scF, out + 16 * 80);

    // RoI extraction -> regions bf16
    {
        long tot = (long)16384 * 432;
        roi_kernel<<<(unsigned)((tot + 255) / 256), 256, 0, stream>>>(images, boxes, regions);
    }

    // region head
    conv(regions, 0, rw1p, rb1, r1out, 0, 16384,   3, 12, 12, 128, 6, 6);
    conv(r1out,   0, rw2p, rb2, lat,   0, 16384, 128,  6,  6, 256, 3, 3);

    // score-weighted reduction + classifier
    {
        dim3 grid((2304 + 255) / 256, 16);
        hidden_kernel<<<grid, 256, 0, stream>>>(scF, lat, hidden);
    }
    logits_kernel<<<16, 128, 0, stream>>>(hidden, ow, ob, out);
}
